// QuadraticConv_31920196943947
// MI455X (gfx1250) — compile-verified, matrix operand order checked by reading, not run on hardware
//
#include <hip/hip_runtime.h>
#include <hip/hip_bf16.h>

// ---------------------------------------------------------------------------
// Quadratic (Volterra) conv as implicit GEMM on CDNA5 WMMA bf16 tensor cores.
//   M = B*H*W = 16384 locations, N = O = 128, K = C*(9 linear + 45 triu pairs)
//   padded to C*64 = 4096.  f32 accumulation.
// B-panel staging goes through the Tensor Data Mover (tensor_load_to_lds,
// 6-arg builtin form on this toolchain); image halo tiles are VALU-staged
// (they need zero-padding); next channel is prefetched (global_prefetch_b8).
// ---------------------------------------------------------------------------

typedef __attribute__((ext_vector_type(16))) __bf16 bf16x16;
typedef __attribute__((ext_vector_type(8)))  __bf16 bf16x8;
typedef __attribute__((ext_vector_type(8)))  float  floatx8;
typedef unsigned int uintx4 __attribute__((ext_vector_type(4)));
typedef int          intx4  __attribute__((ext_vector_type(4)));
typedef int          intx8  __attribute__((ext_vector_type(8)));

#define B_    4
#define C_    64
#define H_    64
#define W_    64
#define O_    128
#define PACKN (C_ * 8 * 2 * 32 * 16)   // 524288 packed bf16 weight elements

#if defined(__has_builtin)
#if __has_builtin(__builtin_amdgcn_tensor_load_to_lds) && \
    __has_builtin(__builtin_amdgcn_s_wait_tensorcnt)
#define HAVE_TDM 1
#endif
#endif
#ifndef HAVE_TDM
#define HAVE_TDM 0
#endif

// ---------------------------------------------------------------------------
// Pack Wl [O,C,3,3] + triu(Wv) [O,C,9,9] into bf16 WMMA-B fragment order:
//   index = ((((c*8 + ntile)*2 + kstep)*32 + lane)*16 + v)
// B-matrix (32K x 16N) 16-bit lane layout: lane&15 = N column,
//   lanes 0-15 hold K {0..7, 16..23}, lanes 16-31 hold K {8..15, 24..31}.
// Per-channel feature index f: 0..8 = patch value k, 9..53 = pair (i<=j),
//   54..63 = zero pad.
// ---------------------------------------------------------------------------
__global__ __launch_bounds__(256) void pack_weights_kernel(
    const float* __restrict__ Wl, const float* __restrict__ Wv,
    __bf16* __restrict__ Bp) {
  int id = blockIdx.x * 256 + threadIdx.x;
  if (id >= PACKN) return;
  int v     = id & 15;
  int lane  = (id >> 4) & 31;
  int kstep = (id >> 9) & 1;
  int ntile = (id >> 10) & 7;
  int c     = id >> 13;
  int n    = ntile * 16 + (lane & 15);
  int kloc = (v < 8 ? v : v + 8) + ((lane >> 4) << 3);
  int f    = kstep * 32 + kloc;
  float val = 0.f;
  if (f < 9) {
    val = Wl[(n * C_ + c) * 9 + f];
  } else if (f < 54) {
    int pf = f - 9, i = 0, base = 0;
    while (pf >= base + 9 - i) { base += 9 - i; ++i; }
    int j = i + (pf - base);
    val = Wv[(n * C_ + c) * 81 + i * 9 + j];
  }
  Bp[id] = (__bf16)val;
}

// ---------------------------------------------------------------------------
// Main kernel: 256 blocks = (b:4) x (rowpair:32) x (o-half:2), 8 wave32s.
// Block tile: M=128 locations (2 image rows x 64 cols), N=64 outputs.
// Each wave owns 16 M-rows x 64 N-cols = 1 A-frag x 4 WMMAs per k-step.
// ---------------------------------------------------------------------------
__global__ __launch_bounds__(256) void volterra_wmma_kernel(
    const float* __restrict__ img, const __bf16* __restrict__ Bp,
    const float* __restrict__ bias, float* __restrict__ out) {
  __shared__ float  imgT[4][72];         // input halo: rows h0-1..h0+2, cols -1..64
  __shared__ __bf16 featT[128][72];      // A tile: 128 locations x 64 feats (+pad)
  __shared__ __bf16 bT[4 * 2 * 32 * 16]; // B panel for this o-half, one channel

  const int tid   = threadIdx.x;
  const int lane  = tid & 31;
  const int wave  = tid >> 5;
  const int nhalf = blockIdx.x & 1;
  const int rp    = (blockIdx.x >> 1) & 31;
  const int b     = blockIdx.x >> 6;
  const int h0    = rp * 2;

  floatx8 acc[4];
#pragma unroll
  for (int ni = 0; ni < 4; ++ni)
#pragma unroll
    for (int r = 0; r < 8; ++r) acc[ni][r] = 0.f;

  const float* imgBase = img + (size_t)b * C_ * H_ * W_;
  const int hclamp = (h0 > 0) ? (h0 - 1) : 0;

  for (int c = 0; c < C_; ++c) {
    __syncthreads();  // previous iter's LDS consumers done

    const float* ic = imgBase + c * H_ * W_;

    // --- stage this block's B panel (8 KB, fragment-swizzled) --------------
#if HAVE_TDM
    if (wave == 0) {
      // Tensor DMA descriptor: 2D tensor, data_size=8B, tile 1024x1 (= 8 KB)
      unsigned ldsAddr = (unsigned)(uintptr_t)(&bT[0]);
      unsigned long long ga =
          (unsigned long long)(uintptr_t)(Bp + (size_t)(c * 8 + nhalf * 4) * 1024);
      uintx4 g0;
      g0[0] = 1u;                                   // count=1 (valid descriptor)
      g0[1] = ldsAddr;                              // lds_addr
      g0[2] = (unsigned)ga;                         // global_addr[31:0]
      g0[3] = (unsigned)(ga >> 32) | 0x80000000u;   // global_addr[56:32] | type=2
      intx8 g1;
      g1[0] = 0x00030000;      // workgroup_mask=0, data_size=3 (8B), no flags
      g1[1] = 0x04000000;      // tensor_dim0 = 1024 (low16 in [31:16])
      g1[2] = 0x00010000;      // tensor_dim0 hi16=0, tensor_dim1 = 1
      g1[3] = 0x04000000;      // tensor_dim1 hi16=0, tile_dim0 = 1024
      g1[4] = 0x00000001;      // tile_dim1 = 1, tile_dim2 = 0
      g1[5] = 1024;            // tensor_dim0_stride low32
      g1[6] = 0;               // stride hi16, tensor_dim1_stride low16
      g1[7] = 0;               // tensor_dim1_stride hi32
      intx4 z4;
      z4[0] = 0; z4[1] = 0; z4[2] = 0; z4[3] = 0;
      intx8 z8;
      z8[0] = 0; z8[1] = 0; z8[2] = 0; z8[3] = 0;
      z8[4] = 0; z8[5] = 0; z8[6] = 0; z8[7] = 0;
      __builtin_amdgcn_tensor_load_to_lds(g0, g1, z4, z4, z8, 0);
      __builtin_amdgcn_s_wait_tensorcnt((short)0);
    }
#else
    {
      const uint4* src = (const uint4*)(Bp + (size_t)(c * 8 + nhalf * 4) * 1024);
      uint4* dst = (uint4*)bT;
      dst[tid]       = src[tid];
      dst[tid + 256] = src[tid + 256];
    }
#endif

    // --- stage image halo tile (4 rows x 66 cols, zero-padded) -------------
    for (int idx = tid; idx < 4 * 66; idx += 256) {
      int r = idx / 66, col = idx % 66 - 1;
      int hin = h0 - 1 + r;
      float vv = 0.f;
      if (hin >= 0 && hin < H_ && col >= 0 && col < W_) vv = ic[hin * W_ + col];
      imgT[r][col + 1] = vv;
    }

    // prefetch next channel's halo region into cache while we do the MMA
    if (c + 1 < C_) {
      __builtin_prefetch(ic + H_ * W_ + hclamp * W_ + tid, 0, 1);
    }
    __syncthreads();

    // --- expand quadratic features into A tile (2 threads per location) ----
    {
      int loc = tid >> 1, half = tid & 1;
      int hl = loc >> 6, w = loc & 63;
      float p[9];
#pragma unroll
      for (int k = 0; k < 9; ++k) p[k] = imgT[hl + k / 3][w + k % 3];
      __bf16* fr = &featT[loc][0];
      if (half == 0) {
#pragma unroll
        for (int k = 0; k < 9; ++k) fr[k] = (__bf16)p[k];
        int f = 9;
#pragma unroll
        for (int i = 0; i < 3; ++i)
#pragma unroll
          for (int j = i; j < 9; ++j) { fr[f] = (__bf16)(p[i] * p[j]); ++f; }
      } else {
        int f = 33;
#pragma unroll
        for (int i = 3; i < 9; ++i)
#pragma unroll
          for (int j = i; j < 9; ++j) { fr[f] = (__bf16)(p[i] * p[j]); ++f; }
#pragma unroll
        for (int k = 54; k < 64; ++k) fr[k] = (__bf16)0.f;
      }
    }
    __syncthreads();

    // --- load ALL fragments first, then issue 8 WMMAs back-to-back ---------
    const int am  = wave * 16 + (lane & 15);
    const int kb0 = (lane >> 4) << 3;     // A layout: lo K-octet per lane half

    bf16x16 afrag[2];
#pragma unroll
    for (int ks = 0; ks < 2; ++ks) {
      bf16x8 alo = *(const bf16x8*)&featT[am][kb0 + ks * 32];
      bf16x8 ahi = *(const bf16x8*)&featT[am][kb0 + ks * 32 + 16];
      afrag[ks] = __builtin_shufflevector(alo, ahi,
          0, 1, 2, 3, 4, 5, 6, 7, 8, 9, 10, 11, 12, 13, 14, 15);
    }
    bf16x16 bfrag[2][4];
#pragma unroll
    for (int ks = 0; ks < 2; ++ks)
#pragma unroll
      for (int ni = 0; ni < 4; ++ni) {
        const __bf16* bp = &bT[((ni * 2 + ks) * 32 + lane) * 16];
        bf16x8 blo = *(const bf16x8*)bp;
        bf16x8 bhi = *(const bf16x8*)(bp + 8);
        bfrag[ks][ni] = __builtin_shufflevector(blo, bhi,
            0, 1, 2, 3, 4, 5, 6, 7, 8, 9, 10, 11, 12, 13, 14, 15);
      }
#pragma unroll
    for (int ks = 0; ks < 2; ++ks)
#pragma unroll
      for (int ni = 0; ni < 4; ++ni)   // ni fastest: consecutive WMMAs use
        acc[ni] = __builtin_amdgcn_wmma_f32_16x16x32_bf16(   // distinct accs
            false, afrag[ks], false, bfrag[ks][ni], (short)0, acc[ni],
            false, false);
  }

  // --- epilogue: C/D layout VGPR r -> M=r(+8 for hi lanes), N=lane&15 ------
  const int lhalf = lane >> 4, ln = lane & 15;
#pragma unroll
  for (int ni = 0; ni < 4; ++ni) {
    int o = nhalf * 64 + ni * 16 + ln;
    float bs = bias[o];
#pragma unroll
    for (int r = 0; r < 8; ++r) {
      int m = wave * 16 + lhalf * 8 + r;   // 0..127 within block tile
      int h = h0 + (m >> 6);
      int w = m & 63;
      out[(((size_t)b * O_ + o) * H_ + h) * W_ + w] = acc[ni][r] + bs;
    }
  }
}

// ---------------------------------------------------------------------------
extern "C" void kernel_launch(void* const* d_in, const int* in_sizes, int n_in,
                              void* d_out, int out_size, void* d_ws, size_t ws_size,
                              hipStream_t stream) {
  const float* img  = (const float*)d_in[0];   // [4,64,64,64]
  const float* wl   = (const float*)d_in[1];   // [128,64,3,3]
  const float* wv   = (const float*)d_in[2];   // [128,64,9,9] (triu)
  const float* bias = (const float*)d_in[3];   // [128]
  __bf16* Bp = (__bf16*)d_ws;                  // 1 MB packed bf16 B-matrix

  pack_weights_kernel<<<PACKN / 256, 256, 0, stream>>>(wl, wv, Bp);
  volterra_wmma_kernel<<<256, 256, 0, stream>>>(img, Bp, bias, (float*)d_out);
}
